// Attention_33217277067420
// MI455X (gfx1250) — matrix operand order repaired and checked by reading, not compiled, not run on hardware
//
#include <hip/hip_runtime.h>
#include <math.h>

// ---------------------------------------------------------------------------
// Attention (qkv 1x1-conv -> l2norm(n) -> scaled softmax attention -> 1x1-conv)
// MI455X / gfx1250: wave32, WMMA 16x16xK, async global->LDS tile staging.
//   - fp32 projections:  v_wmma_f32_16x16x4_f32   (full reference precision)
//   - attention core:    v_wmma_f32_16x16x32_bf16 (f32 accumulate, flash-style)
//   - K/V tiles: global_load_async_to_lds_b128 + s_wait_asynccnt (ASYNCcnt)
// Shapes: B=8, C=256, N=1024, HID=512, H=8, D=64.
// Workspace layout (72 MB needed):
//   [ 0MB) qf   fp32 [B][512][N]      16MB
//   [16MB) kf   fp32 [B][512][N]      16MB
//   [32MB) attnO fp32 [B][512][N]     16MB
//   [48MB) qnT  bf16 [B][H][N][D]      8MB   (A-operand layout for Q^T K)
//   [56MB) kn   bf16 [B][512][N]       8MB   (B-operand layout for Q^T K)
//   [64MB) vT   bf16 [B][H][N][D]      8MB   (B-operand layout for P V^T)
// ---------------------------------------------------------------------------

typedef __attribute__((ext_vector_type(2)))  float  v2f;
typedef __attribute__((ext_vector_type(8)))  float  v8f;
typedef __attribute__((ext_vector_type(16))) __bf16 v16bf;

#define NB   8
#define NC   256
#define NT   1024
#define NHID 512
#define NH   8
#define ND   64

// CDNA5 async copy: 16B per lane, global -> LDS, tracked by ASYNCcnt.
__device__ __forceinline__ void async_copy16(const void* gsrc, void* ldst) {
  const unsigned laddr = (unsigned)(size_t)ldst;      // low 32b = LDS offset
  asm volatile("global_load_async_to_lds_b128 %0, %1, off"
               :: "v"(laddr), "v"(gsrc) : "memory");
}
__device__ __forceinline__ void wait_asynccnt0() {
  asm volatile("s_wait_asynccnt 0" ::: "memory");
}

// ---------------------------------------------------------------------------
// Kernel 1: QKV projection, fp32 WMMA.  D[o,t] = sum_c Wqkv[o,c] * X[b,c,t]
// ---------------------------------------------------------------------------
__global__ __launch_bounds__(256) void qkv_proj_kernel(
    const float* __restrict__ x, const float* __restrict__ w,
    float* __restrict__ qf, float* __restrict__ kf, __bf16* __restrict__ vT) {
  const int lane = threadIdx.x & 31;
  const int wv   = threadIdx.x >> 5;
  const int hi   = lane >> 4;
  const int lr   = lane & 15;
  const int b    = blockIdx.z;
  const int t0   = blockIdx.x * 16;
  const int o0   = (blockIdx.y * 8 + wv) * 16;

  const float* xb = x + (size_t)b * NC * NT;
  v8f acc = {};
  for (int c0 = 0; c0 < NC; c0 += 4) {
    const int ka = c0 + (hi ? 2 : 0);          // A/B 32-bit K mapping (ISA 7.12.2)
    v2f a, bb;
    a[0]  = w[(o0 + lr) * NC + ka];
    a[1]  = w[(o0 + lr) * NC + ka + 1];
    bb[0] = xb[(size_t)ka * NT + t0 + lr];
    bb[1] = xb[(size_t)(ka + 1) * NT + t0 + lr];
    acc = __builtin_amdgcn_wmma_f32_16x16x4_f32(false, a, false, bb,
                                                (short)0, acc, false, false);
  }
#pragma unroll
  for (int r = 0; r < 8; ++r) {
    const int o = o0 + r + (hi ? 8 : 0);       // C/D mapping: m=r+(hi?8:0), n=lr
    const int t = t0 + lr;
    const float val = acc[r];
    if (o < NHID) {
      qf[((size_t)b * NHID + o) * NT + t] = val;
    } else if (o < 2 * NHID) {
      kf[((size_t)b * NHID + (o - NHID)) * NT + t] = val;
    } else {
      const int hd = o - 2 * NHID;
      const int h = hd >> 6, d = hd & 63;
      vT[(((size_t)b * NH + h) * NT + t) * ND + d] = (__bf16)val;
    }
  }
}

// ---------------------------------------------------------------------------
// Kernel 2: l2-normalize q,k over the token axis n.  One block per row.
// ---------------------------------------------------------------------------
__global__ __launch_bounds__(256) void l2norm_kernel(
    const float* __restrict__ qf, const float* __restrict__ kf,
    __bf16* __restrict__ qnT, __bf16* __restrict__ kn) {
  __shared__ float red[8];
  const int row = blockIdx.x;                 // 0..8191 (q rows then k rows)
  const bool isQ = row < NB * NHID;
  const int r = isQ ? row : row - NB * NHID;  // r = b*512 + h*64 + d
  const float* src = (isQ ? qf : kf) + (size_t)r * NT;
  const int tid = threadIdx.x;

  const float4 val = ((const float4*)src)[tid];
  float ss = val.x * val.x + val.y * val.y + val.z * val.z + val.w * val.w;
#pragma unroll
  for (int m = 1; m < 32; m <<= 1) ss += __shfl_xor(ss, m, 32);
  if ((tid & 31) == 0) red[tid >> 5] = ss;
  __syncthreads();
  float tot = 0.f;
#pragma unroll
  for (int i = 0; i < 8; ++i) tot += red[i];
  const float inv = 1.0f / fmaxf(sqrtf(tot), 1e-12f);

  const int tbase = tid * 4;
  const float pv[4] = {val.x * inv, val.y * inv, val.z * inv, val.w * inv};
  if (isQ) {  // write token-major: qnT[bh][t][d]
    const int b = r >> 9, hd = r & 511;
    const int h = hd >> 6, d = hd & 63;
    const size_t base = ((size_t)(b * NH + h) * NT) * ND + d;
#pragma unroll
    for (int i = 0; i < 4; ++i)
      qnT[base + (size_t)(tbase + i) * ND] = (__bf16)pv[i];
  } else {    // kn flat layout identical to kf: [b*512+hd][t]
    const size_t base = (size_t)r * NT + tbase;
#pragma unroll
    for (int i = 0; i < 4; ++i) kn[base + i] = (__bf16)pv[i];
  }
}

// ---------------------------------------------------------------------------
// Kernel 3: fused flash attention per (b,h).  Each wave owns 16 query rows;
// the block cooperatively stages the shared 64x32 K tile and 32x64 V tile in
// LDS via async b128 copies (ASYNCcnt), then 8 bf16 WMMAs per j-block.
// ---------------------------------------------------------------------------
__global__ __launch_bounds__(256) void attn_kernel(
    const __bf16* __restrict__ qnT, const __bf16* __restrict__ kn,
    const __bf16* __restrict__ vT, float* __restrict__ attnO) {
  __shared__ __bf16 kT[64][40];        // K tile [d][jj], row stride 80B (16B mult)
  __shared__ __bf16 vS[32][72];        // V tile [jj][d], row stride 144B (16B mult)
  __shared__ __bf16 plds[8][16][34];   // per-wave P staging (C/D -> A relayout)

  const int tid  = threadIdx.x;
  const int lane = tid & 31;
  const int wv   = tid >> 5;
  const int hi   = lane >> 4;
  const int lr   = lane & 15;
  const int bh   = blockIdx.y;                  // b*8 + h
  const int i0   = blockIdx.x * 128 + wv * 16;  // query-row base for this wave

  // Async-copy slice assignment: 256 threads x 16B cover each 4KB tile.
  const int krow = tid >> 2, kseg = (tid & 3) * 8;   // K: 64 rows x 4 chunks
  const int vrow = tid >> 3, vseg = (tid & 7) * 8;   // V: 32 rows x 8 chunks

  // Preload Q A-fragments (16 rows x 64 d, two 16x32 fragments).
  const __bf16* qrow = qnT + ((size_t)bh * NT + i0 + lr) * ND;
  v16bf qa0, qa1;
#pragma unroll
  for (int h = 0; h < 16; ++h) {
    const int k = (h < 8 ? h : h + 8) + (hi ? 8 : 0);   // A 16x32 K mapping
    qa0[h] = qrow[k];
    qa1[h] = qrow[32 + k];
  }

  v8f acc[4] = {v8f{}, v8f{}, v8f{}, v8f{}};
  float mrow[8], lrow[8];
#pragma unroll
  for (int r = 0; r < 8; ++r) { mrow[r] = -1e30f; lrow[r] = 0.f; }

  const __bf16* kb = kn + (size_t)bh * ND * NT;   // [d][t]
  const __bf16* vb = vT + (size_t)bh * NT * ND;   // [t][d]

  for (int j0 = 0; j0 < NT; j0 += 32) {
    // ---- stage K (64x32) and V (32x64) tiles into LDS (async, b128/lane) ----
    __syncthreads();   // protect previous iteration's LDS reads
    async_copy16(kb + (size_t)krow * NT + j0 + kseg, &kT[krow][kseg]);
    async_copy16(vb + (size_t)(j0 + vrow) * ND + vseg, &vS[vrow][vseg]);
    wait_asynccnt0();
    __syncthreads();

    // ---- S = (Q^T K) for 16 rows x 32 keys (4 bf16 WMMAs) ----
    v8f s0 = {}, s1 = {};
    {
      v16bf k00, k01, k10, k11;
#pragma unroll
      for (int h = 0; h < 16; ++h) {
        const int kk = (h < 8 ? h : h + 8) + (hi ? 8 : 0); // B 32x16 K mapping
        k00[h] = kT[kk][lr];
        k01[h] = kT[32 + kk][lr];
        k10[h] = kT[kk][16 + lr];
        k11[h] = kT[32 + kk][16 + lr];
      }
      s0 = __builtin_amdgcn_wmma_f32_16x16x32_bf16(false, qa0, false, k00, (short)0, s0, false, false);
      s0 = __builtin_amdgcn_wmma_f32_16x16x32_bf16(false, qa1, false, k01, (short)0, s0, false, false);
      s1 = __builtin_amdgcn_wmma_f32_16x16x32_bf16(false, qa0, false, k10, (short)0, s1, false, false);
      s1 = __builtin_amdgcn_wmma_f32_16x16x32_bf16(false, qa1, false, k11, (short)0, s1, false, false);
    }

    // ---- online softmax: rows live across 16 lanes of each half-wave ----
#pragma unroll
    for (int r = 0; r < 8; ++r) {
      const float a = s0[r] * 10.0f;   // SCALE
      const float b = s1[r] * 10.0f;
      float mx = fmaxf(a, b);
#pragma unroll
      for (int m = 1; m < 16; m <<= 1) mx = fmaxf(mx, __shfl_xor(mx, m, 32));
      const float mnew = fmaxf(mrow[r], mx);
      const float p0 = __expf(a - mnew);
      const float p1 = __expf(b - mnew);
      float ps = p0 + p1;
#pragma unroll
      for (int m = 1; m < 16; m <<= 1) ps += __shfl_xor(ps, m, 32);
      const float alpha = __expf(mrow[r] - mnew);
      lrow[r] = lrow[r] * alpha + ps;
      mrow[r] = mnew;
      acc[0][r] *= alpha; acc[1][r] *= alpha;
      acc[2][r] *= alpha; acc[3][r] *= alpha;
      const int mrw = r + (hi ? 8 : 0);                 // C/D row mapping
      plds[wv][mrw][lr]      = (__bf16)p0;
      plds[wv][mrw][16 + lr] = (__bf16)p1;
    }
    asm volatile("s_wait_dscnt 0" ::: "memory");        // per-wave LDS RAW fence

    v16bf pa;                                           // P as 16x32 A-fragment
#pragma unroll
    for (int h = 0; h < 16; ++h) {
      const int k = (h < 8 ? h : h + 8) + (hi ? 8 : 0);
      pa[h] = plds[wv][lr][k];
    }

    // ---- O += P * V  (4 bf16 WMMAs over d) ----
#pragma unroll
    for (int f = 0; f < 4; ++f) {
      v16bf vfr;
#pragma unroll
      for (int h = 0; h < 16; ++h) {
        const int jj = (h < 8 ? h : h + 8) + (hi ? 8 : 0);
        vfr[h] = vS[jj][f * 16 + lr];
      }
      acc[f] = __builtin_amdgcn_wmma_f32_16x16x32_bf16(false, pa, false, vfr,
                                                       (short)0, acc[f], false, false);
    }
  }

  // ---- finalize: divide by row sum, store attnO[b][h*64+d][t] ----
  float* ob = attnO + (size_t)bh * ND * NT;
#pragma unroll
  for (int f = 0; f < 4; ++f) {
#pragma unroll
    for (int r = 0; r < 8; ++r) {
      const int d = f * 16 + lr;
      const int t = i0 + r + (hi ? 8 : 0);
      ob[(size_t)d * NT + t] = acc[f][r] / lrow[r];
    }
  }
}

// ---------------------------------------------------------------------------
// Kernel 4: output projection, fp32 WMMA.  out[o,t] = sum_k Wout[o,k]*A[k,t]+b
// ---------------------------------------------------------------------------
__global__ __launch_bounds__(256) void out_proj_kernel(
    const float* __restrict__ attnO, const float* __restrict__ w_out,
    const float* __restrict__ b_out, float* __restrict__ out) {
  const int lane = threadIdx.x & 31;
  const int wv   = threadIdx.x >> 5;
  const int hi   = lane >> 4;
  const int lr   = lane & 15;
  const int b    = blockIdx.z;
  const int t0   = blockIdx.x * 16;
  const int o0   = (blockIdx.y * 8 + wv) * 16;

  const float* ab = attnO + (size_t)b * NHID * NT;
  v8f acc = {};
  for (int c0 = 0; c0 < NHID; c0 += 4) {
    const int ka = c0 + (hi ? 2 : 0);
    v2f a, bb;
    a[0]  = w_out[(o0 + lr) * NHID + ka];
    a[1]  = w_out[(o0 + lr) * NHID + ka + 1];
    bb[0] = ab[(size_t)ka * NT + t0 + lr];
    bb[1] = ab[(size_t)(ka + 1) * NT + t0 + lr];
    acc = __builtin_amdgcn_wmma_f32_16x16x4_f32(false, a, false, bb,
                                                (short)0, acc, false, false);
  }
#pragma unroll
  for (int r = 0; r < 8; ++r) {
    const int o = o0 + r + (hi ? 8 : 0);
    out[((size_t)b * NC + o) * NT + t0 + lr] = acc[r] + b_out[o];
  }
}

// ---------------------------------------------------------------------------
extern "C" void kernel_launch(void* const* d_in, const int* in_sizes, int n_in,
                              void* d_out, int out_size, void* d_ws, size_t ws_size,
                              hipStream_t stream) {
  (void)in_sizes; (void)n_in; (void)out_size; (void)ws_size;
  const float* x     = (const float*)d_in[0];
  const float* w_qkv = (const float*)d_in[1];
  const float* w_out = (const float*)d_in[2];
  const float* b_out = (const float*)d_in[3];
  float* out = (float*)d_out;

  char* ws = (char*)d_ws;
  float*  qf    = (float*)(ws + (size_t)0);
  float*  kf    = (float*)(ws + ((size_t)16 << 20));
  float*  attnO = (float*)(ws + ((size_t)32 << 20));
  __bf16* qnT   = (__bf16*)(ws + ((size_t)48 << 20));
  __bf16* kn    = (__bf16*)(ws + ((size_t)56 << 20));
  __bf16* vT    = (__bf16*)(ws + ((size_t)64 << 20));

  // 1) QKV projection: 96 o-tiles x 64 t-tiles x 8 batches, 8 waves/block
  qkv_proj_kernel<<<dim3(64, 12, NB), 256, 0, stream>>>(x, w_qkv, qf, kf, vT);
  // 2) l2norm over n for q and k (8192 rows)
  l2norm_kernel<<<dim3(2 * NB * NHID), 256, 0, stream>>>(qf, kf, qnT, kn);
  // 3) fused attention: 8 i-tiles of 128 rows x 64 (b,h) pairs
  attn_kernel<<<dim3(NT / 128, NB * NH), 256, 0, stream>>>(qnT, kn, vT, attnO);
  // 4) output projection + bias
  out_proj_kernel<<<dim3(64, 2, NB), 256, 0, stream>>>(attnO, w_out, b_out, out);
}
